// SocialModel_21500606283978
// MI455X (gfx1250) — matrix operand order repaired and checked by reading, not compiled, not run on hardware
//
#include <hip/hip_runtime.h>
#include <hip/hip_bf16.h>

// CDNA5 / gfx1250 persistent-workgroup Social-LSTM.
// Single 1024-thread workgroup (32 wave32s on one WGP) runs the whole T=128
// recurrence; both per-step GEMMs run on v_wmma_f32_16x16x32_f16 with 1x4
// register blocking (one A fragment feeds 4 WMMAs, reuse_a hint set).

typedef _Float16 h16;
typedef __attribute__((ext_vector_type(16))) _Float16 v16h;
typedef __attribute__((ext_vector_type(8)))  _Float16 v8h;
typedef __attribute__((ext_vector_type(8)))  float    v8f;

#define T_STEPS 128
#define NPED    256
#define RNN     128
#define EMB     64
#define NOUT    5

// ---------------------------------------------------------------------------
// Init: convert weights f32 -> f16 in the WMMA B-fragment (lane-swizzled)
// layout, so each lane loads its 16-half B fragment as one aligned v16h.
//  B1: te path,  [K=128][N=1024], N = g*64+e, B1[k][g*64+e] = W_tens[e][g*128+k]
//  B2: gate path,[K=256][N=512],  B2[k][q]   = k<128 ? W_ih[q][k] : W_hh[q][k-128]
// fragment linear index: ((ct*KT + kt)*32 + lane)*16 + i,
//   k = kt*32 + (lane>=16 ? 16 : 0) + i,  n = ct*16 + (lane&15)
// ---------------------------------------------------------------------------
__global__ void init_weights_kernel(const float* __restrict__ W_tens,
                                    const float* __restrict__ W_ih,
                                    const float* __restrict__ W_hh,
                                    h16* __restrict__ B1, h16* __restrict__ B2) {
  int idx = blockIdx.x * blockDim.x + threadIdx.x;
  if (idx < 131072) {                       // B1: 128*1024 elements
    int e1 = idx;
    int i    = e1 & 15;
    int lane = (e1 >> 4) & 31;
    int kc   = e1 >> 9;
    int kt   = kc & 3;                      // K tiles: 128/32 = 4
    int ct   = kc >> 2;                     // N tiles: 1024/16 = 64
    int k = kt * 32 + ((lane >> 4) * 16) + i;
    int n = ct * 16 + (lane & 15);
    int g = n >> 6, e = n & 63;
    B1[e1] = (h16)W_tens[e * 2048 + g * 128 + k];
  } else if (idx < 262144) {                // B2: 256*512 elements
    int e2 = idx - 131072;
    int i    = e2 & 15;
    int lane = (e2 >> 4) & 31;
    int kc   = e2 >> 9;
    int kt   = kc & 7;                      // K tiles: 256/32 = 8
    int ct   = kc >> 3;                     // N tiles: 512/16 = 32
    int k = kt * 32 + ((lane >> 4) * 16) + i;
    int q = ct * 16 + (lane & 15);
    float v = (k < 128) ? W_ih[q * 128 + k] : W_hh[q * 128 + (k - 128)];
    B2[e2] = (h16)v;
  }
}

// ---------------------------------------------------------------------------
// Main persistent kernel.
// LDS: acat[256][256] f16  (cols 0-63 ie | 64-127 te | 128-255 h)  = 128 KB
//      valid-pair lists, positions, biases                          ~ 24 KB
// Scratch (d_ws): swizzled B1/B2 f16, Mp f32[256][1024],
//                 gates f32[256][512], cstate f32[256][128]
// ---------------------------------------------------------------------------
__global__ __launch_bounds__(1024, 1)
void social_lstm_kernel(const float* __restrict__ V,
                        const float* __restrict__ W_emb,
                        const float* __restrict__ b_emb,
                        const float* __restrict__ b_tens,
                        const float* __restrict__ b_ih,
                        const float* __restrict__ b_hh,
                        const float* __restrict__ W_out,
                        const float* __restrict__ b_out,
                        const h16*  __restrict__ B1,
                        const h16*  __restrict__ B2,
                        float* __restrict__ Mp,
                        float* __restrict__ gbuf,
                        float* __restrict__ cstate,
                        float* __restrict__ out) {
  __shared__ h16            acat[NPED * 256];
  __shared__ unsigned short vlist[NPED * 32];
  __shared__ float          px[NPED], py[NPED];
  __shared__ unsigned int   cnt[NPED];
  __shared__ float          bg[512];
  __shared__ float          sb_emb[EMB], sb_tens[EMB], sW_emb[EMB * 2];
  __shared__ float          sW_out[NOUT * RNN], sb_out[NOUT];

  const int tid  = threadIdx.x;
  const int lane = tid & 31;     // wave32
  const int wave = tid >> 5;     // 32 waves

  // one-time setup (runs every launch -> deterministic)
  for (int i = tid; i < 512; i += 1024)        bg[i] = b_ih[i] + b_hh[i];
  for (int i = tid; i < EMB; i += 1024)        { sb_emb[i] = b_emb[i]; sb_tens[i] = b_tens[i]; }
  for (int i = tid; i < EMB * 2; i += 1024)    sW_emb[i] = W_emb[i];
  for (int i = tid; i < NOUT * RNN; i += 1024) sW_out[i] = W_out[i];
  if (tid < NOUT) sb_out[tid] = b_out[tid];
  for (int i = tid; i < NPED * 256; i += 1024) acat[i] = (h16)0.f;     // h0 = 0
  for (int i = tid; i < NPED * RNN; i += 1024) cstate[i] = 0.f;        // c0 = 0
  __syncthreads();

  const float wb = 32.0f / 720.0f * 2.0f;
  const float hb = 32.0f / 576.0f * 2.0f;

  for (int t = 0; t < T_STEPS; ++t) {
    // ---- phase 1a: load frame positions, zero pair counters ----
    if (tid < NPED) {
      px[tid]  = V[t * NPED + tid];
      py[tid]  = V[T_STEPS * NPED + t * NPED + tid];
      cnt[tid] = 0u;
    }
    __syncthreads();

    // ---- phase 1b: input embedding relu(pos@W_emb.T + b) -> acat[:,0:64] ----
    for (int q = 0; q < 16; ++q) {
      int idx = tid + q * 1024;                 // 256*64
      int n = idx >> 6, e = idx & 63;
      float v = sb_emb[e] + px[n] * sW_emb[e * 2 + 0] + py[n] * sW_emb[e * 2 + 1];
      acat[n * 256 + e] = (h16)fmaxf(v, 0.f);
    }
    // ---- phase 1c: build compacted valid-neighbor lists (~0.8% of pairs) ----
#pragma unroll 2
    for (int q = 0; q < 64; ++q) {
      int p = tid * 64 + q;                     // 256*256 pairs
      int i = p >> 8, j = p & 255;
      if (i == j) continue;
      int cx = (int)floorf((px[j] - px[i] + wb * 0.5f) * (4.0f / wb));
      int cy = (int)floorf((py[j] - py[i] + hb * 0.5f) * (4.0f / hb));
      if (cx < 0 || cx > 3 || cy < 0 || cy > 3) continue;
      unsigned int pos = atomicAdd(&cnt[i], 1u);
      if (pos < 32u)
        vlist[i * 32 + pos] = (unsigned short)(j | ((cx + cy * 4) << 8));
    }
    __syncthreads();

    // ---- phase 2: GEMM1  Mp[256][1024] = h[256][128] x B1[128][1024] ----
    // 1x4 register blocking: 16 row-tiles x 16 col-groups (4 N-tiles each)
    for (int mt = wave; mt < 256; mt += 32) {          // uniform per wave
      int rt = mt >> 4, cg = mt & 15;
      int m0 = rt * 16;
      v8f c0 = {}, c1 = {}, c2 = {}, c3 = {};
      int m = m0 + (lane & 15);
      const h16* arow = &acat[m * 256 + 128];          // h segment
      const int koff = (lane >> 4) * 16;               // contiguous k layout (matches B)
      const v16h* bbase = (const v16h*)B1 + ((size_t)cg * 16 * 32 + lane);
#pragma unroll 1
      for (int kt = 0; kt < 4; ++kt) {
        union { v16h v; v8h h[2]; } a;
        a.h[0] = *(const v8h*)(arow + kt * 32 + koff);
        a.h[1] = *(const v8h*)(arow + kt * 32 + koff + 8);
        v16h b0 = bbase[(0 * 4 + kt) * 32];
        v16h b1 = bbase[(1 * 4 + kt) * 32];
        v16h b2 = bbase[(2 * 4 + kt) * 32];
        v16h b3 = bbase[(3 * 4 + kt) * 32];
        c0 = __builtin_amdgcn_wmma_f32_16x16x32_f16(false, a.v, false, b0, (short)0, c0, true,  false);
        c1 = __builtin_amdgcn_wmma_f32_16x16x32_f16(false, a.v, false, b1, (short)0, c1, true,  false);
        c2 = __builtin_amdgcn_wmma_f32_16x16x32_f16(false, a.v, false, b2, (short)0, c2, true,  false);
        c3 = __builtin_amdgcn_wmma_f32_16x16x32_f16(false, a.v, false, b3, (short)0, c3, false, false);
      }
      int n = cg * 64 + (lane & 15);
      int mrow = m0 + ((lane >> 4) * 8);
#pragma unroll
      for (int v = 0; v < 8; ++v) {
        float* row = &Mp[(mrow + v) * 1024 + n];
        row[0] = c0[v]; row[16] = c1[v]; row[32] = c2[v]; row[48] = c3[v];
      }
    }
    __threadfence();
    __syncthreads();

    // ---- phase 3: sparse gather  te[i,e] = relu(b + sum Mp[j][cell*64+e]) ----
    for (int q = 0; q < 16; ++q) {
      int idx = tid + q * 1024;                 // 256*64
      int i = idx >> 6, e = idx & 63;
      float s = sb_tens[e];
      unsigned int m = cnt[i]; if (m > 32u) m = 32u;
      for (unsigned int u = 0; u < m; ++u) {
        unsigned short ent = vlist[i * 32 + u];
        int j = ent & 255, cell = ent >> 8;
        s += Mp[j * 1024 + cell * 64 + e];
      }
      acat[i * 256 + 64 + e] = (h16)fmaxf(s, 0.f);
    }
    __syncthreads();

    // ---- phase 4: GEMM2  gates[256][512] = acat[256][256] x B2[256][512] ----
    // 16 row-tiles x 8 col-groups (4 N-tiles each)
    for (int mt = wave; mt < 128; mt += 32) {          // uniform per wave
      int rt = mt >> 3, cg = mt & 7;
      int m0 = rt * 16;
      v8f c0 = {}, c1 = {}, c2 = {}, c3 = {};
      int m = m0 + (lane & 15);
      const h16* arow = &acat[m * 256];
      const int koff = (lane >> 4) * 16;
      const v16h* bbase = (const v16h*)B2 + ((size_t)cg * 32 * 32 + lane);
#pragma unroll 1
      for (int kt = 0; kt < 8; ++kt) {
        union { v16h v; v8h h[2]; } a;
        a.h[0] = *(const v8h*)(arow + kt * 32 + koff);
        a.h[1] = *(const v8h*)(arow + kt * 32 + koff + 8);
        v16h b0 = bbase[(0 * 8 + kt) * 32];
        v16h b1 = bbase[(1 * 8 + kt) * 32];
        v16h b2 = bbase[(2 * 8 + kt) * 32];
        v16h b3 = bbase[(3 * 8 + kt) * 32];
        c0 = __builtin_amdgcn_wmma_f32_16x16x32_f16(false, a.v, false, b0, (short)0, c0, true,  false);
        c1 = __builtin_amdgcn_wmma_f32_16x16x32_f16(false, a.v, false, b1, (short)0, c1, true,  false);
        c2 = __builtin_amdgcn_wmma_f32_16x16x32_f16(false, a.v, false, b2, (short)0, c2, true,  false);
        c3 = __builtin_amdgcn_wmma_f32_16x16x32_f16(false, a.v, false, b3, (short)0, c3, false, false);
      }
      int n = cg * 64 + (lane & 15);
      int mrow = m0 + ((lane >> 4) * 8);
#pragma unroll
      for (int v = 0; v < 8; ++v) {
        float* row = &gbuf[(mrow + v) * 512 + n];
        row[0] = c0[v]; row[16] = c1[v]; row[32] = c2[v]; row[48] = c3[v];
      }
    }
    __threadfence();
    __syncthreads();

    // ---- phase 5: LSTM elementwise (f32); thread owns n=tid>>2, 32 r's ----
    {
      int n  = tid >> 2;
      int rb = (tid & 3) * 32;
      const float* gp = &gbuf[n * 512];
#pragma unroll 4
      for (int k = 0; k < 32; ++k) {
        int r = rb + k;
        float ig = gp[r]       + bg[r];
        float fg = gp[128 + r] + bg[128 + r];
        float gg = gp[256 + r] + bg[256 + r];
        float og = gp[384 + r] + bg[384 + r];
        float c_old = cstate[n * 128 + r];
        float si = 1.f / (1.f + __expf(-ig));
        float sf = 1.f / (1.f + __expf(-fg));
        float so = 1.f / (1.f + __expf(-og));
        float cn = sf * c_old + si * tanhf(gg);
        float hn = so * tanhf(cn);
        cstate[n * 128 + r] = cn;
        acat[n * 256 + 128 + r] = (h16)hn;
        if (t == T_STEPS - 1) {
          out[T_STEPS * NPED * NOUT + n * 128 + r] = hn;               // hT
          out[T_STEPS * NPED * NOUT + NPED * RNN + n * 128 + r] = cn;  // cT
        }
      }
    }
    __syncthreads();

    // ---- phase 6: output projection out[t,n,:] = h_new @ W_out.T + b ----
    for (int q = 0; q < 2; ++q) {
      int idx = tid + q * 1024;                 // 256*5 = 1280 outputs
      if (idx >= NPED * NOUT) break;
      int n = idx / NOUT, o = idx % NOUT;
      float s = sb_out[o];
      const v8h* hv = (const v8h*)&acat[n * 256 + 128];   // 16B-aligned
      const float* wo = &sW_out[o * 128];
      for (int rbk = 0; rbk < 16; ++rbk) {
        v8h hh = hv[rbk];
#pragma unroll
        for (int i = 0; i < 8; ++i) s += (float)hh[i] * wo[rbk * 8 + i];
      }
      out[(t * NPED + n) * NOUT + o] = s;
    }
    __syncthreads();
  }
}

extern "C" void kernel_launch(void* const* d_in, const int* in_sizes, int n_in,
                              void* d_out, int out_size, void* d_ws, size_t ws_size,
                              hipStream_t stream) {
  const float* V      = (const float*)d_in[0];
  // d_in[1] = PedsList (identity arange, unused)
  const float* W_emb  = (const float*)d_in[2];
  const float* b_emb  = (const float*)d_in[3];
  const float* W_tens = (const float*)d_in[4];
  const float* b_tens = (const float*)d_in[5];
  const float* W_ih   = (const float*)d_in[6];
  const float* b_ih   = (const float*)d_in[7];
  const float* W_hh   = (const float*)d_in[8];
  const float* b_hh   = (const float*)d_in[9];
  const float* W_out  = (const float*)d_in[10];
  const float* b_out  = (const float*)d_in[11];

  char* ws = (char*)d_ws;
  h16*   B1   = (h16*)(ws);                                   // 256 KB
  h16*   B2   = (h16*)(ws + (256 << 10));                     // 256 KB
  float* Mp   = (float*)(ws + (512 << 10));                   // 1 MB
  float* gbuf = (float*)(ws + (512 << 10) + (1 << 20));       // 512 KB
  float* cst  = (float*)(ws + (512 << 10) + (1 << 20) + (512 << 10)); // 128 KB

  init_weights_kernel<<<262144 / 256, 256, 0, stream>>>(W_tens, W_ih, W_hh, B1, B2);
  social_lstm_kernel<<<1, 1024, 0, stream>>>(V, W_emb, b_emb, b_tens, b_ih, b_hh,
                                             W_out, b_out, B1, B2, Mp, gbuf, cst,
                                             (float*)d_out);
}
